// LRU_31293131719484
// MI455X (gfx1250) — matrix operand order repaired and checked by reading, not compiled, not run on hardware
//
#include <hip/hip_runtime.h>
#include <hip/hip_bf16.h>
#include <math.h>

// ---------------------------------------------------------------------------
// LRU forward for MI455X (gfx1250), wave32 + WMMA bf16 with hi/lo fp32 split.
//   prep : split inputs/Wi/Wo into bf16 hi/lo planes (one streaming pass)
//   GEMM1: u = inputs @ Wi^T + bi              (16384 x 2048, K=1024)
//   scan : x_t = lambda * x_{t-1} + u_t  (complex, per channel), x *= gamma
//   GEMM2: out = relu(xr @ Wo^T + bo)          (16384 x 1024, K=2048)
// GEMM main loop is pure bf16: LDS tiles staged with GLOBAL_LOAD_ASYNC_TO_LDS
// (double-buffered, ASYNCcnt-paced) when the builtin is available.
// ---------------------------------------------------------------------------

typedef __attribute__((ext_vector_type(16))) __bf16 v16bf;
typedef __attribute__((ext_vector_type(4)))  __bf16 v4bf;
typedef __attribute__((ext_vector_type(8)))  float  v8f;
typedef __attribute__((ext_vector_type(4)))  int    v4i;

static_assert(sizeof(v16bf) == 32, "v16bf must be 32 bytes");

union Frag16 {            // one WMMA A/B operand: 16 bf16 = two 16B LDS loads
    v16bf v;
    uint4 q[2];
};

constexpr int BB   = 4;
constexpr int LL   = 4096;
constexpr int DD   = 1024;
constexpr int MTOT = BB * LL;        // 16384
constexpr int NSEG = 16;             // L-segments for the scan
constexpr int LSEG = LL / NSEG;      // 256
constexpr int LDSK = 40;             // 32 k + 8 pad (bank-conflict-free b128)

#if __has_builtin(__builtin_amdgcn_global_load_async_to_lds_b128)
#define HAS_ASYNC_LDS 1
#define GLOBAL_AS __attribute__((address_space(1)))
#define LDS_AS    __attribute__((address_space(3)))
#else
#define HAS_ASYNC_LDS 0
#endif

__device__ __forceinline__ void wait_async_8() {
#if HAS_ASYNC_LDS
#if __has_builtin(__builtin_amdgcn_s_wait_asynccnt)
    __builtin_amdgcn_s_wait_asynccnt(8);
#else
    asm volatile("s_wait_asynccnt 0x8" ::: "memory");
#endif
#endif
}
__device__ __forceinline__ void wait_async_0() {
#if HAS_ASYNC_LDS
#if __has_builtin(__builtin_amdgcn_s_wait_asynccnt)
    __builtin_amdgcn_s_wait_asynccnt(0);
#else
    asm volatile("s_wait_asynccnt 0x0" ::: "memory");
#endif
#endif
}

// Copy one 16B chunk global(bf16) -> LDS(bf16), async when available.
__device__ __forceinline__ void copy16(const __bf16* g, __bf16* l) {
#if HAS_ASYNC_LDS
    __builtin_amdgcn_global_load_async_to_lds_b128(
        (GLOBAL_AS v4i*)g, (LDS_AS v4i*)l, 0, 0);
#else
    *reinterpret_cast<uint4*>(l) = *reinterpret_cast<const uint4*>(g);
#endif
}

// ---------------------------------------------------------------------------
// Elementwise split: f32 -> bf16 hi + bf16 lo planes. n multiple of 1024.
// ---------------------------------------------------------------------------
__global__ __launch_bounds__(256) void split_bf16_kernel(
    const float* __restrict__ in, __bf16* __restrict__ hi,
    __bf16* __restrict__ lo, int n)
{
    const int i4 = (blockIdx.x * 256 + threadIdx.x) * 4;
    if (i4 >= n) return;
    float4 f = *reinterpret_cast<const float4*>(in + i4);
    float fv[4] = {f.x, f.y, f.z, f.w};
    v4bf hv, lv;
#pragma unroll
    for (int j = 0; j < 4; ++j) {
        __bf16 h = (__bf16)fv[j];
        hv[j] = h;
        lv[j] = (__bf16)(fv[j] - (float)h);
    }
    *reinterpret_cast<v4bf*>(hi + i4) = hv;
    *reinterpret_cast<v4bf*>(lo + i4) = lv;
}

// ---------------------------------------------------------------------------
// Pure-bf16 GEMM: C[m,n] = act( sum_k A[m,k]*B[n,k] + bias[n] ), with A,B
// pre-split into hi/lo planes; product = Ah*Bh + Ah*Bl + Al*Bh (~fp32).
// Block: 256 threads (8 waves), 128x128 C tile, wave tile 64(M) x 32(N).
// Double-buffered LDS (80 KB), async global->LDS staging.
// ---------------------------------------------------------------------------
template <bool RELU>
__global__ __launch_bounds__(256) void gemm_bf16_kernel(
    const __bf16* __restrict__ Ah, const __bf16* __restrict__ Al,
    const __bf16* __restrict__ Bh, const __bf16* __restrict__ Bl,
    const float* __restrict__ bias, float* __restrict__ Cout,
    int M, int N, int K)
{
    __shared__ __bf16 sAh[2][128 * LDSK];
    __shared__ __bf16 sAl[2][128 * LDSK];
    __shared__ __bf16 sBh[2][128 * LDSK];
    __shared__ __bf16 sBl[2][128 * LDSK];

    const int tid  = threadIdx.x;
    const int lane = tid & 31;
    const int wv   = tid >> 5;
    const int hl   = lane >> 4;   // lane half (0/1)
    const int lr   = lane & 15;   // row/col within half

    const int m0 = blockIdx.y * 128;
    const int n0 = blockIdx.x * 128;
    const int wm = (wv >> 2) * 64;   // 2 waves along M
    const int wn = (wv & 3) * 32;    // 4 waves along N

    v8f acc[4][2];
#pragma unroll
    for (int mt = 0; mt < 4; ++mt)
#pragma unroll
        for (int nt = 0; nt < 2; ++nt)
            acc[mt][nt] = v8f{};

    // per-thread copy slots: 512 16B-chunks per tile / 256 threads = 2 each
    const int crow0 = tid >> 2;              // 0..63
    const int crow1 = crow0 + 64;            // 64..127
    const int ck8   = (tid & 3) * 8;         // 0,8,16,24

    auto copy_chunk = [&](int kc, int buf) {
        const size_t ga0 = (size_t)(m0 + crow0) * K + kc + ck8;
        const size_t ga1 = (size_t)(m0 + crow1) * K + kc + ck8;
        const size_t gb0 = (size_t)(n0 + crow0) * K + kc + ck8;
        const size_t gb1 = (size_t)(n0 + crow1) * K + kc + ck8;
        const int l0 = crow0 * LDSK + ck8;
        const int l1 = crow1 * LDSK + ck8;
        copy16(Ah + ga0, &sAh[buf][l0]);
        copy16(Ah + ga1, &sAh[buf][l1]);
        copy16(Al + ga0, &sAl[buf][l0]);
        copy16(Al + ga1, &sAl[buf][l1]);
        copy16(Bh + gb0, &sBh[buf][l0]);
        copy16(Bh + gb1, &sBh[buf][l1]);
        copy16(Bl + gb0, &sBl[buf][l0]);
        copy16(Bl + gb1, &sBl[buf][l1]);
    };

    auto compute_chunk = [&](int buf) {
        const uint4* pAh = reinterpret_cast<const uint4*>(&sAh[buf][0]);
        const uint4* pAl = reinterpret_cast<const uint4*>(&sAl[buf][0]);
        const uint4* pBh = reinterpret_cast<const uint4*>(&sBh[buf][0]);
        const uint4* pBl = reinterpret_cast<const uint4*>(&sBl[buf][0]);

        // B fragments (32x16 KxN): lane col = lr; lanes 0-15 hold K 0..15,
        // lanes 16-31 hold K 16..31.
        Frag16 bh[2], bl[2];
#pragma unroll
        for (int nt = 0; nt < 2; ++nt) {
            const int nrow = wn + nt * 16 + lr;
            const int e0 = nrow * LDSK + hl * 16;
            bh[nt].q[0] = pBh[e0 / 8];
            bh[nt].q[1] = pBh[e0 / 8 + 1];
            bl[nt].q[0] = pBl[e0 / 8];
            bl[nt].q[1] = pBl[e0 / 8 + 1];
        }
        // A fragments (16x32): lane row = lr; lanes 0-15: K 0..15;
        // lanes 16-31: K 16..31.
#pragma unroll
        for (int mt = 0; mt < 4; ++mt) {
            const int mrow = wm + mt * 16 + lr;
            const int a0 = mrow * LDSK + hl * 16;
            const int a1 = mrow * LDSK + hl * 16 + 8;
            Frag16 fah, fal;
            fah.q[0] = pAh[a0 / 8];
            fah.q[1] = pAh[a1 / 8];
            fal.q[0] = pAl[a0 / 8];
            fal.q[1] = pAl[a1 / 8];
#pragma unroll
            for (int nt = 0; nt < 2; ++nt) {
                acc[mt][nt] = __builtin_amdgcn_wmma_f32_16x16x32_bf16(
                    false, fah.v, false, bh[nt].v, (short)0, acc[mt][nt], false, false);
                acc[mt][nt] = __builtin_amdgcn_wmma_f32_16x16x32_bf16(
                    false, fah.v, false, bl[nt].v, (short)0, acc[mt][nt], false, false);
                acc[mt][nt] = __builtin_amdgcn_wmma_f32_16x16x32_bf16(
                    false, fal.v, false, bh[nt].v, (short)0, acc[mt][nt], false, false);
            }
        }
    };

    const int nchunks = K / 32;
    copy_chunk(0, 0);
    for (int ic = 0; ic < nchunks - 1; ++ic) {
        copy_chunk((ic + 1) * 32, (ic + 1) & 1);   // prefetch next
        wait_async_8();                            // current buffer complete
        __syncthreads();
        compute_chunk(ic & 1);
        __syncthreads();                           // all waves done reading
    }
    wait_async_0();
    __syncthreads();
    compute_chunk((nchunks - 1) & 1);

    // epilogue: bias (+relu); C layout: vgpr vg -> row (vg + 8*half), col lr
#pragma unroll
    for (int nt = 0; nt < 2; ++nt) {
        const int n = n0 + wn + nt * 16 + lr;
        const float bn = bias[n];
#pragma unroll
        for (int mt = 0; mt < 4; ++mt) {
#pragma unroll
            for (int vg = 0; vg < 8; ++vg) {
                const int m = m0 + wm + mt * 16 + hl * 8 + vg;
                float v = acc[mt][nt][vg] + bn;
                if (RELU) v = v > 0.0f ? v : 0.0f;
                Cout[(size_t)m * N + n] = v;
            }
        }
    }
}

// ---------------------------------------------------------------------------
// complex helpers
// ---------------------------------------------------------------------------
__device__ __forceinline__ float2 cmul(float2 a, float2 b) {
    return make_float2(a.x * b.x - a.y * b.y, a.x * b.y + a.y * b.x);
}
__device__ __forceinline__ float2 cmad(float2 lam, float2 x, float2 u) {
    float2 r = cmul(lam, x);
    return make_float2(r.x + u.x, r.y + u.y);
}

// Scan pass A: per (b, channel, segment) local scan; emit segment-end state.
__global__ __launch_bounds__(256) void lru_scan_ends(
    const float* __restrict__ u, const float* __restrict__ plog,
    float2* __restrict__ E)
{
    const int tid = threadIdx.x;
    const int seg = blockIdx.x;            // 0..NSEG-1
    const int b   = blockIdx.y >> 2;       // 0..3
    const int cg  = blockIdx.y & 3;        // 0..3 (D/256 groups)
    const int c   = cg * 256 + tid;

    const float v  = expf(plog[c]);
    const float th = expf(plog[DD + c]);
    const float el = expf(-v);
    const float2 lam = make_float2(el * cosf(th), el * sinf(th));

    float2 x = make_float2(0.0f, 0.0f);
    size_t base = ((size_t)b * LL + (size_t)seg * LSEG) * (2 * DD) + 2 * c;
    for (int l = 0; l < LSEG; ++l) {
        float2 uv = *reinterpret_cast<const float2*>(u + base + (size_t)l * (2 * DD));
        x = cmad(lam, x, uv);
    }
    E[((size_t)b * DD + c) * NSEG + seg] = x;
}

// Scan pass B: serial combine over NSEG segments per channel with lambda^LSEG.
__global__ __launch_bounds__(256) void lru_scan_carries(
    const float2* __restrict__ E, const float* __restrict__ plog,
    float2* __restrict__ Carr)
{
    const int idx = blockIdx.x * 256 + threadIdx.x;   // b*D + c
    const int c   = idx & (DD - 1);
    const float v  = expf(plog[c]);
    const float th = expf(plog[DD + c]);
    const float mag = expf(-(float)LSEG * v);
    const float ang = (float)LSEG * th;
    const float2 lamL = make_float2(mag * cosf(ang), mag * sinf(ang));

    float2 F = make_float2(0.0f, 0.0f);
    for (int s = 0; s < NSEG; ++s) {
        Carr[(size_t)idx * NSEG + s] = F;             // carry *into* segment s
        float2 e = E[(size_t)idx * NSEG + s];
        F = cmad(lamL, F, e);
    }
}

// Scan pass C: redo local scan seeded with carry, apply gamma, emit xr
// pre-split into bf16 hi/lo planes (GEMM2 A layout: [m][k], k = c or D+c).
__global__ __launch_bounds__(256) void lru_scan_apply(
    const float* __restrict__ u, const float* __restrict__ plog,
    const float2* __restrict__ Carr,
    __bf16* __restrict__ Xh, __bf16* __restrict__ Xl)
{
    const int tid = threadIdx.x;
    const int seg = blockIdx.x;
    const int b   = blockIdx.y >> 2;
    const int cg  = blockIdx.y & 3;
    const int c   = cg * 256 + tid;

    const float v  = expf(plog[c]);
    const float th = expf(plog[DD + c]);
    const float g  = expf(plog[2 * DD + c]);
    const float el = expf(-v);
    const float2 lam = make_float2(el * cosf(th), el * sinf(th));

    float2 x = Carr[((size_t)b * DD + c) * NSEG + seg];
    size_t ubase = ((size_t)b * LL + (size_t)seg * LSEG) * (2 * DD) + 2 * c;
    size_t obase = ((size_t)b * LL + (size_t)seg * LSEG) * (2 * DD) + c;
    for (int l = 0; l < LSEG; ++l) {
        float2 uv = *reinterpret_cast<const float2*>(u + ubase + (size_t)l * (2 * DD));
        x = cmad(lam, x, uv);
        const float yr = g * x.x;
        const float yi = g * x.y;
        __bf16 hr = (__bf16)yr;  __bf16 hi = (__bf16)yi;
        __bf16 lrr = (__bf16)(yr - (float)hr);
        __bf16 lii = (__bf16)(yi - (float)hi);
        size_t o = obase + (size_t)l * (2 * DD);
        Xh[o]      = hr;   Xh[o + DD] = hi;
        Xl[o]      = lrr;  Xl[o + DD] = lii;
    }
}

// ---------------------------------------------------------------------------
extern "C" void kernel_launch(void* const* d_in, const int* in_sizes, int n_in,
                              void* d_out, int out_size, void* d_ws, size_t ws_size,
                              hipStream_t stream) {
    (void)in_sizes; (void)n_in; (void)out_size; (void)ws_size;

    const float* inputs = (const float*)d_in[0];   // (4,4096,1024)
    const float* Wi     = (const float*)d_in[1];   // (2048,1024)
    const float* bi     = (const float*)d_in[2];   // (2048)
    const float* Wo     = (const float*)d_in[3];   // (1024,2048)
    const float* bo     = (const float*)d_in[4];   // (1024)
    const float* plog   = (const float*)d_in[5];   // (3,1024)
    float* out          = (float*)d_out;           // (4,4096,1024)

    // workspace layout
    char* w = (char*)d_ws;
    size_t off = 0;
    float*  u    = (float*)(w + off);  off += (size_t)MTOT * 2 * DD * sizeof(float);   // 128 MB
    __bf16* Xh   = (__bf16*)(w + off); off += (size_t)MTOT * 2 * DD * sizeof(__bf16);  // 64 MB
    __bf16* Xl   = (__bf16*)(w + off); off += (size_t)MTOT * 2 * DD * sizeof(__bf16);  // 64 MB
    __bf16* inH  = (__bf16*)(w + off); off += (size_t)MTOT * DD * sizeof(__bf16);      // 32 MB
    __bf16* inL  = (__bf16*)(w + off); off += (size_t)MTOT * DD * sizeof(__bf16);      // 32 MB
    __bf16* WiH  = (__bf16*)(w + off); off += (size_t)2 * DD * DD * sizeof(__bf16);    // 4 MB
    __bf16* WiL  = (__bf16*)(w + off); off += (size_t)2 * DD * DD * sizeof(__bf16);
    __bf16* WoH  = (__bf16*)(w + off); off += (size_t)2 * DD * DD * sizeof(__bf16);
    __bf16* WoL  = (__bf16*)(w + off); off += (size_t)2 * DD * DD * sizeof(__bf16);
    float2* E    = (float2*)(w + off); off += (size_t)BB * DD * NSEG * sizeof(float2);
    float2* Carr = (float2*)(w + off);

    // prep: split fp32 operands into bf16 hi/lo planes
    {
        const int nIn = MTOT * DD;          // 16.7M
        const int nW  = 2 * DD * DD;        // 2.1M
        split_bf16_kernel<<<nIn / 1024, 256, 0, stream>>>(inputs, inH, inL, nIn);
        split_bf16_kernel<<<nW / 1024, 256, 0, stream>>>(Wi, WiH, WiL, nW);
        split_bf16_kernel<<<nW / 1024, 256, 0, stream>>>(Wo, WoH, WoL, nW);
    }

    // GEMM1: u = inputs @ Wi^T + bi     (M=16384, N=2048, K=1024)
    {
        dim3 grid(2 * DD / 128, MTOT / 128);
        gemm_bf16_kernel<false><<<grid, 256, 0, stream>>>(
            inH, inL, WiH, WiL, bi, u, MTOT, 2 * DD, DD);
    }

    // scan
    {
        dim3 grid(NSEG, BB * (DD / 256));
        lru_scan_ends<<<grid, 256, 0, stream>>>(u, plog, E);
        lru_scan_carries<<<BB * DD / 256, 256, 0, stream>>>(E, plog, Carr);
        lru_scan_apply<<<grid, 256, 0, stream>>>(u, plog, Carr, Xh, Xl);
    }

    // GEMM2: out = relu(xr @ Wo^T + bo) (M=16384, N=1024, K=2048)
    {
        dim3 grid(DD / 128, MTOT / 128);
        gemm_bf16_kernel<true><<<grid, 256, 0, stream>>>(
            Xh, Xl, WoH, WoL, bo, out, MTOT, DD, 2 * DD);
    }
}
